// CoeusBlockOptimized_13297218748585
// MI455X (gfx1250) — compile-verified, matrix operand order checked by reading, not run on hardware
//
#include <hip/hip_runtime.h>

// ---------------------------------------------------------------------------
// CDNA5 (gfx1250) implementation of the CoeusBlock forward pass.
// All GEMM-shaped math runs through v_wmma_f32_16x16x32_f16 (wave32 WMMA,
// f16 in / f32 accumulate). Tile staging uses the Tensor Data Mover
// (tensor_load_to_lds + s_wait_tensorcnt) when the toolchain exposes it,
// with a manual global->LDS fallback otherwise. Attention is a flash-style
// causal kernel keeping the running softmax in the WMMA C-layout.
// ---------------------------------------------------------------------------

typedef __attribute__((ext_vector_type(16))) _Float16 v16h;
typedef __attribute__((ext_vector_type(8)))  float    v8f;
typedef unsigned int u32x4 __attribute__((ext_vector_type(4)));
typedef int          i32x8 __attribute__((ext_vector_type(8)));
typedef int          i32x4 __attribute__((ext_vector_type(4)));

#if __has_builtin(__builtin_amdgcn_tensor_load_to_lds) && \
    __has_builtin(__builtin_amdgcn_s_wait_tensorcnt)
#define COEUS_HAVE_TDM 1
#if __has_include(<hip/amd_detail/amd_gfx1250_TDM.h>)
#define COEUS_TDM_ARITY6 1   // therock-10.0 headers -> 6-arg builtin
#endif
#endif

constexpr int  Bc   = 2;
constexpr int  Tc   = 2048;
constexpr int  Dc   = 1024;
constexpr int  Hc   = 16;
constexpr int  HDc  = 64;
constexpr int  Mc   = 256;     // mem compression dim
constexpr int  Sc   = 256;     // T / FREQ memory slots
constexpr int  HIDc = 3072;
constexpr long NRc  = (long)Bc * Tc;   // 4096 rows

#ifdef COEUS_HAVE_TDM
// ---------------------------------------------------------------------------
// Tensor Data Mover: load a 2-D f16 tile [tile_rows x tile_cols] from global
// (row stride stride_elems) into LDS at lds_addr, with LDS row padding
// expressed via the D# pad fields (pad_interval_code per ISA: 0=2,..,3=16,
// 4=32 DWORDs; pad_amount_code: n -> n+1 DWORDs). Descriptor per
// CDNA5 ISA (groups 2/3 zero => 2-D tile).
// ---------------------------------------------------------------------------
__device__ __forceinline__ void tdm_load_tile_f16(
    unsigned int lds_addr, const _Float16* gp, unsigned int tile_cols,
    unsigned int tile_rows, unsigned int stride_elems,
    unsigned int pad_interval_code, unsigned int pad_amount_code) {
  unsigned long long ga = (unsigned long long)gp;
  u32x4 g0 = {};
  g0[0] = 1u;                                   // count=1, user descriptor
  g0[1] = lds_addr;                             // LDS byte address
  g0[2] = (unsigned int)ga;                     // global_addr[31:0]
  g0[3] = (unsigned int)(ga >> 32) | (2u << 30);// global_addr[56:32] | type=2
  const unsigned int td0 = 0x7fffffffu;         // tensor dims: effectively no OOB
  const unsigned int td1 = 0x7fffffffu;
  i32x8 g1 = {};
  g1[0] = (int)(0x00010000u                     // data_size = 2 bytes
                | (1u << 20)                    // pad_enable
                | (pad_interval_code << 22)
                | (pad_amount_code << 25));
  g1[1] = (int)((td0 & 0xffffu) << 16);         // tensor_dim0 lo16
  g1[2] = (int)((td0 >> 16) | ((td1 & 0xffffu) << 16));
  g1[3] = (int)((td1 >> 16) | (tile_cols << 16));   // tile_dim0
  g1[4] = (int)(tile_rows & 0xffffu);               // tile_dim1 ; tile_dim2=0
  g1[5] = (int)stride_elems;                        // tensor_dim0_stride lo32
  g1[6] = 0;
  g1[7] = 0;
  i32x4 gz = {};
#ifdef COEUS_TDM_ARITY6
  i32x8 gz8 = {};
  __builtin_amdgcn_tensor_load_to_lds(g0, g1, gz, gz, gz8, 0);
#else
  __builtin_amdgcn_tensor_load_to_lds(g0, g1, gz, gz, 0);
#endif
}
#endif  // COEUS_HAVE_TDM

// ---------------------------------------------------------------------------
// block-wide reductions (blockDim.x == 256, 8 wave32 waves)
// ---------------------------------------------------------------------------
__device__ __forceinline__ float warp_sum(float v) {
#pragma unroll
  for (int m = 16; m; m >>= 1) v += __shfl_xor(v, m, 32);
  return v;
}
__device__ __forceinline__ float warp_max(float v) {
#pragma unroll
  for (int m = 16; m; m >>= 1) v = fmaxf(v, __shfl_xor(v, m, 32));
  return v;
}
__device__ __forceinline__ float block_sum(float v, float* sm) {
  v = warp_sum(v);
  int w = threadIdx.x >> 5, l = threadIdx.x & 31, nw = blockDim.x >> 5;
  if (l == 0) sm[w] = v;
  __syncthreads();
  float r = (l < nw) ? sm[l] : 0.f;
  r = warp_sum(r);
  __syncthreads();
  return r;
}
__device__ __forceinline__ float block_max(float v, float* sm) {
  v = warp_max(v);
  int w = threadIdx.x >> 5, l = threadIdx.x & 31, nw = blockDim.x >> 5;
  if (l == 0) sm[w] = v;
  __syncthreads();
  float r = (l < nw) ? sm[l] : -3.0e38f;
  r = warp_max(r);
  __syncthreads();
  return r;
}

// ---------------------------------------------------------------------------
// Generic WMMA GEMM:  C[M,N] = A[M,K] * op(B)
//   transB == 0 : B is [N,K] row-major (C = A @ B^T, the "weight" case)
//   transB == 1 : B is [K,N] row-major (C = A @ B)
// Tile: 128x64, 256 threads = 8 waves; wave w owns rows [16w,16w+16) x 64 cols.
// Requires M%128==0, N%64==0, K%32==0 (true for every call below).
// Fragment packing follows the CDNA5 16-bit A-matrix VGPR layout
// (CDNA5 ISA §7.12.2); B is packed mirror-symmetrically.
// A/B tiles are staged by the TDM when available (pad codes 3/3 ->
// 16 data DWORDs + 4 pad DWORDs per row == the 40-half LDS pitch).
// ---------------------------------------------------------------------------
__global__ __launch_bounds__(256)
void gemm_wmma_kernel(const _Float16* __restrict__ A,
                      const _Float16* __restrict__ Bm,
                      float* __restrict__ C,
                      int K, int lda, int ldb, int ldc,
                      long sAb, long sBb, long sCb,
                      int transB) {
  __shared__ _Float16 As[128][40];   // 40-half pitch keeps 16B alignment
  __shared__ _Float16 Bs[64][40];

  A  += (long)blockIdx.z * sAb;
  Bm += (long)blockIdx.z * sBb;
  C  += (long)blockIdx.z * sCb;

  const int tid  = threadIdx.x;
  const int lane = tid & 31;
  const int wid  = tid >> 5;          // 8 waves
  const int r    = lane & 15;         // row-in-tile (A) / col-in-tile (B,C)
  const int kb   = (lane >> 4) << 3;  // K sub-block select per half-wave
  const int row0 = blockIdx.x * 128;
  const int col0 = blockIdx.y * 64;
  const int wrow = wid << 4;

#ifdef COEUS_HAVE_TDM
  const unsigned int asAddr = (unsigned int)(unsigned long long)&As[0][0];
  const unsigned int bsAddr = (unsigned int)(unsigned long long)&Bs[0][0];
#endif

  v8f acc[4] = {};

  for (int k0 = 0; k0 < K; k0 += 32) {
    __syncthreads();
#ifdef COEUS_HAVE_TDM
    // ---- TDM staging: wave 0 issues tensor loads, waits TENSORcnt ------
    if (wid == 0) {
      tdm_load_tile_f16(asAddr, A + (long)row0 * lda + k0, 32, 128,
                        (unsigned int)lda, 3, 3);
      if (!transB)
        tdm_load_tile_f16(bsAddr, Bm + (long)col0 * ldb + k0, 32, 64,
                          (unsigned int)ldb, 3, 3);
    }
    if (transB) {                      // transpose scatter stays manual
      int kk = tid >> 3;               // 0..31
      int ng = (tid & 7) << 3;         // 0..56
      float4 tmp = *(const float4*)(Bm + (long)(k0 + kk) * ldb + col0 + ng);
      const _Float16* tp = (const _Float16*)&tmp;
#pragma unroll
      for (int j = 0; j < 8; ++j) Bs[ng + j][kk] = tp[j];
    }
    if (wid == 0) __builtin_amdgcn_s_wait_tensorcnt(0);
#else
    // ---- manual staging fallback ---------------------------------------
    {
      int i0 = tid << 4;
      int ar = i0 >> 5, ac = i0 & 31;
      const _Float16* src = A + (long)(row0 + ar) * lda + k0 + ac;
      *(float4*)&As[ar][ac]     = *(const float4*)src;
      *(float4*)&As[ar][ac + 8] = *(const float4*)(src + 8);
    }
    if (!transB) {
      int i0 = tid << 3;
      int bn = i0 >> 5, bc = i0 & 31;
      *(float4*)&Bs[bn][bc] =
          *(const float4*)(Bm + (long)(col0 + bn) * ldb + k0 + bc);
    } else {
      int kk = tid >> 3;
      int ng = (tid & 7) << 3;
      float4 tmp = *(const float4*)(Bm + (long)(k0 + kk) * ldb + col0 + ng);
      const _Float16* tp = (const _Float16*)&tmp;
#pragma unroll
      for (int j = 0; j < 8; ++j) Bs[ng + j][kk] = tp[j];
    }
#endif
    __syncthreads();

    // ---- prefetch next K tile (global_prefetch_b8) ----------------------
    if (k0 + 32 < K) {
      __builtin_prefetch(A + (long)(row0 + (tid >> 1)) * lda + k0 + 32 +
                             ((tid & 1) << 4), 0, 1);
    }

    // ---- pack A fragment -------------------------------------------------
    v16h a;
#pragma unroll
    for (int v = 0; v < 8; ++v) {
      int kk = ((v >> 2) << 4) + kb + ((v & 3) << 1);
      a[2 * v]     = As[wrow + r][kk];
      a[2 * v + 1] = As[wrow + r][kk + 1];
    }
    // ---- 4 col-tiles of WMMA --------------------------------------------
#pragma unroll
    for (int ct = 0; ct < 4; ++ct) {
      v16h bfr;
#pragma unroll
      for (int v = 0; v < 8; ++v) {
        int kk = ((v >> 2) << 4) + kb + ((v & 3) << 1);
        bfr[2 * v]     = Bs[ct * 16 + r][kk];
        bfr[2 * v + 1] = Bs[ct * 16 + r][kk + 1];
      }
      acc[ct] = __builtin_amdgcn_wmma_f32_16x16x32_f16(
          false, a, false, bfr, (short)0, acc[ct], false, false);
    }
  }

  // ---- epilogue: C/D layout (VGPR j <-> row j(+8), lane <-> col) ---------
#pragma unroll
  for (int ct = 0; ct < 4; ++ct)
#pragma unroll
    for (int v = 0; v < 8; ++v) {
      int m = v + ((lane >> 4) << 3);
      C[(long)(row0 + wrow + m) * ldc + col0 + ct * 16 + r] = acc[ct][v];
    }
}

// ---------------------------------------------------------------------------
// Flash attention (causal). Grid: (T/64, B*H), 128 threads = 4 waves,
// wave w owns 16 query rows. K/V tiles of 64 keys staged in LDS (TDM when
// available: 64-half rows = 32 data DWORDs + 4 pad DWORDs -> 72-half pitch).
// ---------------------------------------------------------------------------
__global__ __launch_bounds__(128)
void flash_attn_kernel(const _Float16* __restrict__ Q,
                       const _Float16* __restrict__ Kg,
                       const _Float16* __restrict__ Vg,
                       _Float16* __restrict__ O) {
  __shared__ _Float16 Ks[64][72];
  __shared__ _Float16 Vs[64][72];
  __shared__ _Float16 Ps[4][16][72];

  const int tid  = threadIdx.x;
  const int lane = tid & 31;
  const int wid  = tid >> 5;
  const int r    = lane & 15;
  const int hl   = lane >> 4;
  const int kb   = hl << 3;
  const int qt   = blockIdx.x;
  const int b    = blockIdx.y >> 4;
  const int h    = blockIdx.y & 15;
  const int wrow = wid << 4;

#ifdef COEUS_HAVE_TDM
  const unsigned int ksAddr = (unsigned int)(unsigned long long)&Ks[0][0];
  const unsigned int vsAddr = (unsigned int)(unsigned long long)&Vs[0][0];
#endif

  // preload Q fragments (HD=64 -> two K-steps)
  const long qbase = (((long)b * Tc + qt * 64 + wrow + r) * Hc + h) * HDc;
  v16h qa[2];
#pragma unroll
  for (int ks = 0; ks < 2; ++ks)
#pragma unroll
    for (int v = 0; v < 8; ++v) {
      int kk = ks * 32 + ((v >> 2) << 4) + kb + ((v & 3) << 1);
      qa[ks][2 * v]     = Q[qbase + kk];
      qa[ks][2 * v + 1] = Q[qbase + kk + 1];
    }

  v8f o[4] = {};
  float rmax[8], rsum[8];
#pragma unroll
  for (int v = 0; v < 8; ++v) { rmax[v] = -3.0e38f; rsum[v] = 0.f; }

  for (int kt = 0; kt <= qt; ++kt) {
    __syncthreads();
#ifdef COEUS_HAVE_TDM
    if (wid == 0) {
      long src = (((long)b * Tc + kt * 64) * Hc + h) * HDc;
      tdm_load_tile_f16(ksAddr, Kg + src, 64, 64, Hc * HDc, 4, 3);
      tdm_load_tile_f16(vsAddr, Vg + src, 64, 64, Hc * HDc, 4, 3);
      __builtin_amdgcn_s_wait_tensorcnt(0);
    }
#else
#pragma unroll
    for (int j = 0; j < 4; ++j) {
      int i0 = (tid * 4 + j) * 8;
      int rr = i0 >> 6, cc = i0 & 63;
      long src = (((long)b * Tc + kt * 64 + rr) * Hc + h) * HDc + cc;
      *(float4*)&Ks[rr][cc] = *(const float4*)(Kg + src);
      *(float4*)&Vs[rr][cc] = *(const float4*)(Vg + src);
    }
#endif
    __syncthreads();

    // S = Q @ K^T  (16x64 per wave)
    v8f sf[4] = {};
#pragma unroll
    for (int ks = 0; ks < 2; ++ks)
#pragma unroll
      for (int ct = 0; ct < 4; ++ct) {
        v16h bfr;
#pragma unroll
        for (int v = 0; v < 8; ++v) {
          int kk = ks * 32 + ((v >> 2) << 4) + kb + ((v & 3) << 1);
          bfr[2 * v]     = Ks[ct * 16 + r][kk];
          bfr[2 * v + 1] = Ks[ct * 16 + r][kk + 1];
        }
        sf[ct] = __builtin_amdgcn_wmma_f32_16x16x32_f16(
            false, qa[ks], false, bfr, (short)0, sf[ct], false, false);
      }

    // causal mask + online softmax (rows live in half-wave lane groups)
    float sv[4][8], tmax[8];
#pragma unroll
    for (int v = 0; v < 8; ++v) tmax[v] = -3.0e38f;
    const int rowg0 = qt * 64 + wrow + (hl << 3);
#pragma unroll
    for (int ct = 0; ct < 4; ++ct) {
      const int colg = kt * 64 + ct * 16 + r;
#pragma unroll
      for (int v = 0; v < 8; ++v) {
        float val = sf[ct][v] * 0.125f;           // 1/sqrt(HD)
        if (colg > rowg0 + v) val = -1.0e30f;
        sv[ct][v] = val;
        tmax[v] = fmaxf(tmax[v], val);
      }
    }
#pragma unroll
    for (int v = 0; v < 8; ++v) {
#pragma unroll
      for (int m = 8; m; m >>= 1) tmax[v] = fmaxf(tmax[v], __shfl_xor(tmax[v], m, 16));
      float nm   = fmaxf(rmax[v], tmax[v]);
      float corr = __expf(rmax[v] - nm);
      rmax[v] = nm;
      float ps = 0.f;
#pragma unroll
      for (int ct = 0; ct < 4; ++ct) {
        float p = __expf(sv[ct][v] - nm);
        sv[ct][v] = p;
        ps += p;
      }
#pragma unroll
      for (int m = 8; m; m >>= 1) ps += __shfl_xor(ps, m, 16);
      rsum[v] = rsum[v] * corr + ps;
#pragma unroll
      for (int ct = 0; ct < 4; ++ct) o[ct][v] *= corr;
    }

    // P -> LDS (per-wave region), then O += P @ V
#pragma unroll
    for (int ct = 0; ct < 4; ++ct)
#pragma unroll
      for (int v = 0; v < 8; ++v)
        Ps[wid][(hl << 3) + v][ct * 16 + r] = (_Float16)sv[ct][v];
    __syncthreads();

#pragma unroll
    for (int ks = 0; ks < 2; ++ks) {
      v16h pa;
#pragma unroll
      for (int v = 0; v < 8; ++v) {
        int kk = ks * 32 + ((v >> 2) << 4) + kb + ((v & 3) << 1);
        pa[2 * v]     = Ps[wid][r][kk];
        pa[2 * v + 1] = Ps[wid][r][kk + 1];
      }
#pragma unroll
      for (int ct = 0; ct < 4; ++ct) {
        v16h bfr;
#pragma unroll
        for (int v = 0; v < 8; ++v) {
          int kk = ks * 32 + ((v >> 2) << 4) + kb + ((v & 3) << 1);
          bfr[2 * v]     = Vs[kk][ct * 16 + r];
          bfr[2 * v + 1] = Vs[kk + 1][ct * 16 + r];
        }
        o[ct] = __builtin_amdgcn_wmma_f32_16x16x32_f16(
            false, pa, false, bfr, (short)0, o[ct], false, false);
      }
    }
  }

  // epilogue: O / rowsum, written as f16 [B,T,H*HD]
#pragma unroll
  for (int v = 0; v < 8; ++v) {
    float inv = 1.0f / rsum[v];
    const long ob = (((long)b * Tc + qt * 64 + wrow + (hl << 3) + v) * Hc + h) * HDc;
#pragma unroll
    for (int ct = 0; ct < 4; ++ct)
      O[ob + ct * 16 + r] = (_Float16)(o[ct][v] * inv);
  }
}

// ---------------------------------------------------------------------------
// elementwise / reduction glue kernels
// ---------------------------------------------------------------------------
__global__ void cvt_f16_kernel(const float* __restrict__ in,
                               _Float16* __restrict__ out, long n) {
  long i = (long)blockIdx.x * blockDim.x + threadIdx.x;
  if (i < n) out[i] = (_Float16)in[i];
}

__global__ void cvt_f16_strided_kernel(const float* __restrict__ in,
                                       _Float16* __restrict__ out,
                                       int out_stride, int col_ofs, long n) {
  long i = (long)blockIdx.x * blockDim.x + threadIdx.x;
  if (i >= n) return;
  long rr = i >> 10;        // cols_in == 1024
  int  cc = (int)(i & 1023);
  out[rr * out_stride + col_ofs + cc] = (_Float16)in[i];
}

// RMSNorm over D=1024, one block per row; optional f32 and f16 outputs.
__global__ __launch_bounds__(256)
void rms_kernel(const float* __restrict__ x, const float* __restrict__ w,
                float* __restrict__ o32, _Float16* __restrict__ o16) {
  __shared__ float sred[32];
  long row = blockIdx.x;
  const float* xr = x + row * Dc;
  float v[4], ss = 0.f;
#pragma unroll
  for (int j = 0; j < 4; ++j) {
    v[j] = xr[threadIdx.x + j * 256];
    ss += v[j] * v[j];
  }
  ss = block_sum(ss, sred);
  float sc = rsqrtf(ss * (1.0f / Dc) + 1e-6f);
#pragma unroll
  for (int j = 0; j < 4; ++j) {
    int c = threadIdx.x + j * 256;
    float o = v[j] * sc * w[c];
    if (o32) o32[row * Dc + c] = o;
    if (o16) o16[row * Dc + c] = (_Float16)o;
  }
}

// rotary + f32->f16 for Q and K. one thread per (b,t,h,pair)
__global__ void rope_kernel(const float* __restrict__ q, const float* __restrict__ k,
                            const float* __restrict__ fc, const float* __restrict__ fs,
                            _Float16* __restrict__ Qr, _Float16* __restrict__ Kr) {
  long i = (long)blockIdx.x * blockDim.x + threadIdx.x;
  if (i >= (long)Bc * Tc * Hc * (HDc / 2)) return;
  int p  = (int)(i & 31);
  long i2 = i >> 5;
  int hh = (int)(i2 & 15);
  long i3 = i2 >> 4;
  int t  = (int)(i3 & (Tc - 1));
  int b  = (int)(i3 >> 11);
  long base = (((long)b * Tc + t) * Hc + hh) * HDc;
  float c = fc[(long)t * 32 + p], s = fs[(long)t * 32 + p];
  float qr = q[base + 2 * p], qi = q[base + 2 * p + 1];
  float kr = k[base + 2 * p], ki = k[base + 2 * p + 1];
  Qr[base + 2 * p]     = (_Float16)(qr * c - qi * s);
  Qr[base + 2 * p + 1] = (_Float16)(qr * s + qi * c);
  Kr[base + 2 * p]     = (_Float16)(kr * c - ki * s);
  Kr[base + 2 * p + 1] = (_Float16)(kr * s + ki * c);
}

// gate = sigmoid(gpre + b); u = silu(combined[:, :D])
__global__ void rnn_prep_kernel(const float* __restrict__ comb,
                                float* __restrict__ gate_io,
                                const float* __restrict__ gb,
                                float* __restrict__ u) {
  long i = (long)blockIdx.x * blockDim.x + threadIdx.x;
  if (i >= NRc * Dc) return;
  long rr = i >> 10;
  int  d  = (int)(i & 1023);
  gate_io[i] = 1.0f / (1.0f + __expf(-(gate_io[i] + gb[d])));
  float uv = comb[rr * (2 * Dc) + d];
  u[i] = uv / (1.0f + __expf(-uv));
}

// sequential linear-recurrence scan: one thread per (b,d) channel
__global__ void rnn_scan_kernel(const float* __restrict__ gate,
                                const float* __restrict__ u,
                                float* __restrict__ hstate) {
  int c = blockIdx.x * blockDim.x + threadIdx.x;
  if (c >= Bc * Dc) return;
  int b = c >> 10, d = c & 1023;
  long base = (long)b * Tc * Dc + d;
  float s = 0.f;
  for (int t = 0; t < Tc; ++t) {
    long idx = base + (long)t * Dc;
    s = gate[idx] * s + u[idx];
    hstate[idx] = s;
  }
}

// row softmax over 256 cols with pre-scale; f16 output
__global__ __launch_bounds__(256)
void softmax256_kernel(const float* __restrict__ sc, _Float16* __restrict__ out,
                       float scale) {
  __shared__ float sred[32];
  long row = blockIdx.x;
  float v = sc[row * 256 + threadIdx.x] * scale;
  float m = block_max(v, sred);
  float e = __expf(v - m);
  float s = block_sum(e, sred);
  out[row * 256 + threadIdx.x] = (_Float16)(e / s);
}

// out_episodic: mo2 *= sigmoid(dot(h_row, gw) + gb)   (in place)
__global__ __launch_bounds__(256)
void epi_gate_kernel(const float* __restrict__ h, const float* __restrict__ gw,
                     const float* __restrict__ gb, float* __restrict__ mo2) {
  __shared__ float sred[32];
  long row = blockIdx.x;
  float s = 0.f;
#pragma unroll
  for (int j = 0; j < 4; ++j) {
    int c = threadIdx.x + j * 256;
    s += h[row * Dc + c] * gw[c];
  }
  s = block_sum(s, sred);
  float g = 1.0f / (1.0f + __expf(-(s + gb[0])));
#pragma unroll
  for (int j = 0; j < 4; ++j) {
    int c = threadIdx.x + j * 256;
    mo2[row * Dc + c] *= g;
  }
}

// t1 = gemm_out + b1 ; LayerNorm ; silu -> f16
__global__ __launch_bounds__(256)
void ln_silu_kernel(const float* __restrict__ g, const float* __restrict__ b1,
                    const float* __restrict__ lnw, const float* __restrict__ lnb,
                    _Float16* __restrict__ out) {
  __shared__ float sred[32];
  long row = blockIdx.x;
  float v[4], s = 0.f;
#pragma unroll
  for (int j = 0; j < 4; ++j) {
    int c = threadIdx.x + j * 256;
    v[j] = g[row * Dc + c] + b1[c];
    s += v[j];
  }
  float mu = block_sum(s, sred) * (1.0f / Dc);
  float s2 = 0.f;
#pragma unroll
  for (int j = 0; j < 4; ++j) { float d = v[j] - mu; s2 += d * d; }
  float var = block_sum(s2, sred) * (1.0f / Dc);
  float inv = rsqrtf(var + 1e-5f);
#pragma unroll
  for (int j = 0; j < 4; ++j) {
    int c = threadIdx.x + j * 256;
    float ln = (v[j] - mu) * inv * lnw[c] + lnb[c];
    out[row * Dc + c] = (_Float16)(ln / (1.0f + __expf(-ln)));
  }
}

// x1 = x + (1-g2)*out_global + g2*out_local + episodic
__global__ void mix_kernel(const float* __restrict__ x, const float* __restrict__ g2pre,
                           const float* __restrict__ b2, const float* __restrict__ og,
                           const float* __restrict__ ol, const float* __restrict__ epi,
                           float* __restrict__ x1) {
  long i = (long)blockIdx.x * blockDim.x + threadIdx.x;
  if (i >= NRc * Dc) return;
  int d = (int)(i & 1023);
  float g2 = 1.0f / (1.0f + __expf(-(g2pre[i] + b2[d])));
  x1[i] = x[i] + (1.0f - g2) * og[i] + g2 * ol[i] + epi[i];
}

__global__ void silu_mul_kernel(const float* __restrict__ ta, const float* __restrict__ tb,
                                _Float16* __restrict__ out, long n) {
  long i = (long)blockIdx.x * blockDim.x + threadIdx.x;
  if (i >= n) return;
  float a = ta[i];
  out[i] = (_Float16)((a / (1.0f + __expf(-a))) * tb[i]);
}

__global__ void add_inplace_kernel(float* __restrict__ acc, const float* __restrict__ inc, long n) {
  long i = (long)blockIdx.x * blockDim.x + threadIdx.x;
  if (i < n) acc[i] += inc[i];
}

__global__ void final_add_kernel(const float* __restrict__ a, const float* __restrict__ b,
                                 float* __restrict__ out, long n) {
  long i = (long)blockIdx.x * blockDim.x + threadIdx.x;
  if (i < n) out[i] = a[i] + b[i];
}

// ---------------------------------------------------------------------------
// host orchestration
// ---------------------------------------------------------------------------
extern "C" void kernel_launch(void* const* d_in, const int* in_sizes, int n_in,
                              void* d_out, int out_size, void* d_ws, size_t ws_size,
                              hipStream_t stream) {
  (void)in_sizes; (void)n_in; (void)out_size; (void)ws_size;

  const float* x          = (const float*)d_in[0];
  const float* fcos       = (const float*)d_in[1];
  const float* fsin       = (const float*)d_in[2];
  const float* attn_nw    = (const float*)d_in[3];
  const float* wq         = (const float*)d_in[4];
  const float* wk         = (const float*)d_in[5];
  const float* wv         = (const float*)d_in[6];
  const float* wo         = (const float*)d_in[7];
  const float* rnn_gw     = (const float*)d_in[8];
  const float* rnn_gb     = (const float*)d_in[9];
  const float* rnn_inw    = (const float*)d_in[10];
  const float* rnn_outw   = (const float*)d_in[11];
  const float* rnn_cnw    = (const float*)d_in[12];
  const float* mem_compw  = (const float*)d_in[13];
  const float* mem_qw     = (const float*)d_in[14];
  const float* mem_kw     = (const float*)d_in[15];
  const float* mem_vw     = (const float*)d_in[16];
  const float* mem_gw     = (const float*)d_in[17];
  const float* mem_gb     = (const float*)d_in[18];
  const float* lg_rnnw    = (const float*)d_in[19];
  const float* lg_attnw   = (const float*)d_in[20];
  const float* lg_w1      = (const float*)d_in[21];
  const float* lg_b1      = (const float*)d_in[22];
  const float* lg_lnw     = (const float*)d_in[23];
  const float* lg_lnb     = (const float*)d_in[24];
  const float* lg_w2      = (const float*)d_in[25];
  const float* lg_b2      = (const float*)d_in[26];
  const float* ffn_nw     = (const float*)d_in[27];
  const float* r_w1       = (const float*)d_in[28];
  const float* r_w2       = (const float*)d_in[29];
  const float* r_w3       = (const float*)d_in[30];
  const float* r_nw       = (const float*)d_in[31];

  // ---- workspace bump allocator ----
  char* base = (char*)d_ws;
  size_t off = 0;
  auto alloc = [&](size_t bytes) -> char* {
    char* p = base + off;
    off = (off + bytes + 255) & ~(size_t)255;
    return p;
  };
  const long ND  = NRc * Dc;          // 4 M
  const long N2D = NRc * 2 * Dc;      // 8 M
  const long NH  = NRc * HIDc;        // 12 M

  // f16 weight mirrors
  _Float16 *wq16 = (_Float16*)alloc(Dc*Dc*2),  *wk16 = (_Float16*)alloc(Dc*Dc*2);
  _Float16 *wv16 = (_Float16*)alloc(Dc*Dc*2),  *wo16 = (_Float16*)alloc(Dc*Dc*2);
  _Float16 *rgw16 = (_Float16*)alloc(Dc*Dc*2), *riw16 = (_Float16*)alloc(2L*Dc*Dc*2);
  _Float16 *row16 = (_Float16*)alloc(Dc*Dc*2);
  _Float16 *mcw16 = (_Float16*)alloc((long)Mc*Dc*2), *mqw16 = (_Float16*)alloc((long)Mc*Dc*2);
  _Float16 *mkw16 = (_Float16*)alloc((long)Mc*Mc*2), *mvw16 = (_Float16*)alloc((long)Dc*Mc*2);
  _Float16 *lrw16 = (_Float16*)alloc(Dc*Dc*2), *law16 = (_Float16*)alloc(Dc*Dc*2);
  _Float16 *lw1_16 = (_Float16*)alloc(2L*Dc*Dc*2), *lw2_16 = (_Float16*)alloc(Dc*Dc*2);
  _Float16 *rw1_16 = (_Float16*)alloc((long)HIDc*Dc*2);
  _Float16 *rw2_16 = (_Float16*)alloc((long)HIDc*Dc*2);
  _Float16 *rw3_16 = (_Float16*)alloc((long)HIDc*Dc*2);

  // activations
  float    *h32    = (float*)alloc(ND*4);
  _Float16 *h16    = (_Float16*)alloc(ND*2);
  float    *pool   = (float*)alloc(NH*2*4);      // q/k/v then ta/tb
  float    *qf = pool, *kf = pool + ND, *vf = pool + 2*ND;
  _Float16 *Qr   = (_Float16*)alloc(ND*2), *Kr = (_Float16*)alloc(ND*2);
  _Float16 *Vf   = (_Float16*)alloc(ND*2);
  _Float16 *att16 = (_Float16*)alloc(ND*2);
  float    *outl32 = (float*)alloc(ND*4);
  float    *comb32 = (float*)alloc(N2D*4);
  float    *gate32 = (float*)alloc(ND*4);
  float    *u32    = (float*)alloc(ND*4);
  float    *hst32  = (float*)alloc(ND*4);
  _Float16 *hst16  = (_Float16*)alloc(ND*2);
  float    *outg32 = (float*)alloc(ND*4);
  _Float16 *outg16 = (_Float16*)alloc(ND*2);
  _Float16 *outl16 = (_Float16*)alloc(ND*2);
  float    *memk32 = (float*)alloc((long)Bc*Sc*Mc*4);
  _Float16 *memk16 = (_Float16*)alloc((long)Bc*Sc*Mc*2);
  float    *Qm32   = (float*)alloc(NRc*Mc*4);
  _Float16 *Qm16   = (_Float16*)alloc(NRc*Mc*2);
  float    *K2_32  = (float*)alloc((long)Bc*Sc*Mc*4);
  _Float16 *K2_16  = (_Float16*)alloc((long)Bc*Sc*Mc*2);
  float    *msc32  = (float*)alloc(NRc*Sc*4);
  _Float16 *mat16  = (_Float16*)alloc(NRc*Sc*2);
  float    *mo32   = (float*)alloc(ND*4);
  _Float16 *mo16   = (_Float16*)alloc(ND*2);
  float    *mom32  = (float*)alloc(NRc*Mc*4);
  _Float16 *mom16  = (_Float16*)alloc(NRc*Mc*2);
  float    *mo2_32 = (float*)alloc(ND*4);
  _Float16 *di16   = (_Float16*)alloc(N2D*2);
  float    *t1g32  = (float*)alloc(ND*4);
  _Float16 *sln16  = (_Float16*)alloc(ND*2);
  float    *g2p32  = (float*)alloc(ND*4);
  float    *x1_32  = (float*)alloc(ND*4);
  float    *rs32   = (float*)alloc(ND*4);
  _Float16 *n16    = (_Float16*)alloc(ND*2);
  _Float16 *ffn16  = (_Float16*)alloc(NH*2);
  float    *ffo32  = (float*)alloc(ND*4);

  auto cvt = [&](const float* src, _Float16* dst, long n) {
    cvt_f16_kernel<<<(unsigned)((n + 255) / 256), 256, 0, stream>>>(src, dst, n);
  };
  auto gemm = [&](const _Float16* A, const _Float16* Bm, float* C,
                  long Mr, long Nc, int K, int lda, int ldb, int ldc,
                  long sA, long sB, long sC, int batches, int transB) {
    dim3 g((unsigned)(Mr / 128), (unsigned)(Nc / 64), (unsigned)batches);
    gemm_wmma_kernel<<<g, 256, 0, stream>>>(A, Bm, C, K, lda, ldb, ldc,
                                            sA, sB, sC, transB);
  };

  // ---- weight conversions ----
  cvt(wq, wq16, (long)Dc*Dc);       cvt(wk, wk16, (long)Dc*Dc);
  cvt(wv, wv16, (long)Dc*Dc);       cvt(wo, wo16, (long)Dc*Dc);
  cvt(rnn_gw, rgw16, (long)Dc*Dc);  cvt(rnn_inw, riw16, 2L*Dc*Dc);
  cvt(rnn_outw, row16, (long)Dc*Dc);
  cvt(mem_compw, mcw16, (long)Mc*Dc); cvt(mem_qw, mqw16, (long)Mc*Dc);
  cvt(mem_kw, mkw16, (long)Mc*Mc);    cvt(mem_vw, mvw16, (long)Dc*Mc);
  cvt(lg_rnnw, lrw16, (long)Dc*Dc);   cvt(lg_attnw, law16, (long)Dc*Dc);
  cvt(lg_w1, lw1_16, 2L*Dc*Dc);       cvt(lg_w2, lw2_16, (long)Dc*Dc);
  cvt(r_w1, rw1_16, (long)HIDc*Dc);   cvt(r_w2, rw2_16, (long)HIDc*Dc);
  cvt(r_w3, rw3_16, (long)HIDc*Dc);

  // ---- 1) h = rms(x) ----
  rms_kernel<<<(unsigned)NRc, 256, 0, stream>>>(x, attn_nw, h32, h16);

  // ---- 2) q,k,v projections ----
  gemm(h16, wq16, qf, NRc, Dc, Dc, Dc, Dc, Dc, 0, 0, 0, 1, 0);
  gemm(h16, wk16, kf, NRc, Dc, Dc, Dc, Dc, Dc, 0, 0, 0, 1, 0);
  gemm(h16, wv16, vf, NRc, Dc, Dc, Dc, Dc, Dc, 0, 0, 0, 1, 0);

  // ---- 3) rotary + f16 convert ----
  {
    long n = (long)Bc * Tc * Hc * (HDc / 2);
    rope_kernel<<<(unsigned)((n + 255) / 256), 256, 0, stream>>>(qf, kf, fcos, fsin, Qr, Kr);
    cvt(vf, Vf, ND);
  }

  // ---- 4) flash attention ----
  flash_attn_kernel<<<dim3(Tc / 64, Bc * Hc), 128, 0, stream>>>(Qr, Kr, Vf, att16);
  gemm(att16, wo16, outl32, NRc, Dc, Dc, Dc, Dc, Dc, 0, 0, 0, 1, 0);

  // ---- 5) gated linear RNN ----
  gemm(h16, riw16, comb32, NRc, 2 * Dc, Dc, Dc, Dc, 2 * Dc, 0, 0, 0, 1, 0);
  gemm(h16, rgw16, gate32, NRc, Dc, Dc, Dc, Dc, Dc, 0, 0, 0, 1, 0);
  rnn_prep_kernel<<<(unsigned)((ND + 255) / 256), 256, 0, stream>>>(comb32, gate32, rnn_gb, u32);
  rnn_scan_kernel<<<(Bc * Dc + 255) / 256, 256, 0, stream>>>(gate32, u32, hst32);
  rms_kernel<<<(unsigned)NRc, 256, 0, stream>>>(hst32, rnn_cnw, nullptr, hst16);
  gemm(hst16, row16, outg32, NRc, Dc, Dc, Dc, Dc, Dc, 0, 0, 0, 1, 0);

  // ---- 6) episodic memory (slot gather via lda=8*D) ----
  gemm(h16, mcw16, memk32, Sc, Mc, Dc, 8 * Dc, Dc, Mc,
       (long)Tc * Dc, 0, (long)Sc * Mc, Bc, 0);
  cvt(memk32, memk16, (long)Bc * Sc * Mc);
  gemm(h16, mqw16, Qm32, NRc, Mc, Dc, Dc, Dc, Mc, 0, 0, 0, 1, 0);
  cvt(Qm32, Qm16, NRc * Mc);
  gemm(memk16, mkw16, K2_32, (long)Bc * Sc, Mc, Mc, Mc, Mc, Mc, 0, 0, 0, 1, 0);
  cvt(K2_32, K2_16, (long)Bc * Sc * Mc);
  gemm(Qm16, K2_16, msc32, Tc, Sc, Mc, Mc, Mc, Sc,
       (long)Tc * Mc, (long)Sc * Mc, (long)Tc * Sc, Bc, 0);
  softmax256_kernel<<<(unsigned)NRc, 256, 0, stream>>>(msc32, mat16, 0.0625f); // 1/sqrt(M)
  // mo = attn @ mem_slots  (B operand = strided rows of h16, transB)
  gemm(mat16, h16, mo32, Tc, Dc, Sc, Sc, 8 * Dc, Dc,
       (long)Tc * Sc, (long)Tc * Dc, (long)Tc * Dc, Bc, 1);
  cvt(mo32, mo16, ND);
  gemm(mo16, mcw16, mom32, NRc, Mc, Dc, Dc, Dc, Mc, 0, 0, 0, 1, 0);
  cvt(mom32, mom16, NRc * Mc);
  gemm(mom16, mvw16, mo2_32, NRc, Dc, Mc, Mc, Mc, Dc, 0, 0, 0, 1, 0);
  epi_gate_kernel<<<(unsigned)NRc, 256, 0, stream>>>(h32, mem_gw, mem_gb, mo2_32);

  // ---- 7) local/global mixing ----
  cvt(outg32, outg16, ND);
  cvt(outl32, outl16, ND);
  gemm(outg16, lrw16, gate32, NRc, Dc, Dc, Dc, Dc, Dc, 0, 0, 0, 1, 0); // h_rnn (reuse)
  gemm(outl16, law16, u32,    NRc, Dc, Dc, Dc, Dc, Dc, 0, 0, 0, 1, 0); // h_attn (reuse)
  cvt_f16_strided_kernel<<<(unsigned)((ND + 255) / 256), 256, 0, stream>>>(gate32, di16, 2 * Dc, 0, ND);
  cvt_f16_strided_kernel<<<(unsigned)((ND + 255) / 256), 256, 0, stream>>>(u32,    di16, 2 * Dc, Dc, ND);
  gemm(di16, lw1_16, t1g32, NRc, Dc, 2 * Dc, 2 * Dc, 2 * Dc, Dc, 0, 0, 0, 1, 0);
  ln_silu_kernel<<<(unsigned)NRc, 256, 0, stream>>>(t1g32, lg_b1, lg_lnw, lg_lnb, sln16);
  gemm(sln16, lw2_16, g2p32, NRc, Dc, Dc, Dc, Dc, Dc, 0, 0, 0, 1, 0);
  mix_kernel<<<(unsigned)((ND + 255) / 256), 256, 0, stream>>>(x, g2p32, lg_b2,
                                                               outg32, outl32, mo2_32, x1_32);

  // ---- 8) recurrent FFN (2 iters) ----
  rms_kernel<<<(unsigned)NRc, 256, 0, stream>>>(x1_32, ffn_nw, rs32, nullptr);
  float* ta = pool;
  float* tb = pool + NH;
  for (int it = 0; it < 2; ++it) {
    rms_kernel<<<(unsigned)NRc, 256, 0, stream>>>(rs32, r_nw, nullptr, n16);
    gemm(n16, rw1_16, ta, NRc, HIDc, Dc, Dc, Dc, HIDc, 0, 0, 0, 1, 0);
    gemm(n16, rw3_16, tb, NRc, HIDc, Dc, Dc, Dc, HIDc, 0, 0, 0, 1, 0);
    silu_mul_kernel<<<(unsigned)((NH + 255) / 256), 256, 0, stream>>>(ta, tb, ffn16, NH);
    gemm(ffn16, rw2_16, ffo32, NRc, Dc, HIDc, HIDc, HIDc, Dc, 0, 0, 0, 1, 0);
    add_inplace_kernel<<<(unsigned)((ND + 255) / 256), 256, 0, stream>>>(rs32, ffo32, ND);
  }

  // ---- 9) output ----
  final_add_kernel<<<(unsigned)((ND + 255) / 256), 256, 0, stream>>>(x1_32, rs32, (float*)d_out, ND);
}
